// KANLinearFFT_75428215652964
// MI455X (gfx1250) — matrix operand order repaired and checked by reading, not compiled, not run on hardware
//
#include <hip/hip_runtime.h>
#include <hip/hip_bf16.h>

// ---------------------------------------------------------------------------
// KANLinearFFT on MI455X (gfx1250):
//   out[b,o] = sum_i silu(x[b,i]) * scale_base[o,i]
//            + sum_i sum_g cos((g+1)x)*w0[o,i,g] + sin((g+1)x)*w1[o,i,g]
// One fp16 WMMA GEMM: M=4096, N=1024, K=17*1024, channels c = {silu,
// cos1..8, sin1..8}, k = c*1024 + i. Weights packed per launch into fp16
// W[o][k] in d_ws (34 MB, fully L2-resident at 192 MB).
//
// Round-3 changes (latency hiding — round 2 stalled on s_wait_loadcnt 0x0
// directly before every WMMA):
//  * Explicit double-buffered B pipeline: chunk j+1's 16 global_load_b128
//    are in flight while chunk j's 8 WMMAs execute (channel order
//    silu, cos1, sin1, ..., cos8, sin8 so recurrence state advances in step).
//  * saddr addressing: uniform SGPR segment pointer (Wh + i0) + constant
//    per-lane u32 column offsets; channel delta folds into the load imm.
//    Kills the per-fragment v_add_nc_u64 pairs.
//  * silu via v_rcp_f32; harmonics via packed-FMA angle-addition recurrence
//    (only cos(x),sin(x) touch the TRANS pipe that WMMA shares).
// ---------------------------------------------------------------------------

typedef __attribute__((ext_vector_type(16))) _Float16 v16h;
typedef __attribute__((ext_vector_type(8)))  float    v8f;

#define IN_DIM   1024
#define OUT_DIM  1024
#define BATCH    4096
#define GSIZE    8
#define NC       17                 // 1 + 2*GSIZE channels per input dim
#define KDIM     (NC * IN_DIM)      // 17408
#define INV_2PI  0.15915494309189535f

// ---- prep: fold scale_base/scale_spline/coeff into fp16 W[o][k], k=c*1024+i
__global__ __launch_bounds__(256)
void kan_pack_weights(const float* __restrict__ scale_base,
                      const float* __restrict__ scale_spline,
                      const float* __restrict__ coeff,
                      _Float16* __restrict__ Wh) {
  int idx = blockIdx.x * blockDim.x + threadIdx.x;   // 0 .. O*I-1
  int o = idx >> 10;                                 // / IN_DIM
  int i = idx & (IN_DIM - 1);
  float sb = scale_base[idx];
  float ss = scale_spline[idx];
  const float* c0 = coeff + ((size_t)o * IN_DIM + i) * GSIZE;               // t=0
  const float* c1 = coeff + ((size_t)(OUT_DIM + o) * IN_DIM + i) * GSIZE;   // t=1
  _Float16* wrow = Wh + (size_t)o * KDIM + i;
  wrow[0] = (_Float16)sb;                            // channel 0: base path
#pragma unroll
  for (int g = 0; g < GSIZE; ++g) {
    wrow[(size_t)(1 + g) * IN_DIM]         = (_Float16)(ss * c0[g]);  // cos k=g+1
    wrow[(size_t)(1 + GSIZE + g) * IN_DIM] = (_Float16)(ss * c1[g]);  // sin k=g+1
  }
}

// ---- main GEMM: 8 waves/block; wave = 16 rows x 128 cols; block tile 128x128
__global__ __launch_bounds__(256)
void kan_fft_wmma(const float* __restrict__ x,
                  const _Float16* __restrict__ Wh,
                  float* __restrict__ out) {
  const int lane   = threadIdx.x & 31;
  const int wave   = threadIdx.x >> 5;
  const int lane15 = lane & 15;
  const bool hi    = lane >= 16;

  const int m0 = blockIdx.y * 128 + wave * 16;   // rows owned by this wave
  const int n0 = blockIdx.x * 128;               // cols owned by this block

  v8f acc[8] = {};                               // 8 N-subtiles of 16x16 f32

  // A-fragment lane mapping used here (contiguous k, same order as B):
  //   lanes 0-15  hold K = 0..15, lanes 16-31 hold K = 16..31
  const int row  = m0 + lane15;
  const int ka0 = hi ? 16 : 0;    // A-side k offset of this lane's first chunk
  const int kb0 = hi ? 16 : 0;    // B-side k offset (lanes 16-31 hold K 16..31)

  // Per-lane u32 element offsets for the 8 B columns (saddr + voffset form);
  // max (1023*17408 + 16) elements -> byte offset < 2^31.
  unsigned boff[8];
#pragma unroll
  for (int t = 0; t < 8; ++t)
    boff[t] = (unsigned)(n0 + t * 16 + lane15) * (unsigned)KDIM + (unsigned)kb0;
  const unsigned xoff = (unsigned)row * (unsigned)IN_DIM + (unsigned)ka0;

  v16h bf[2][8];                                 // double-buffered B fragments

  // ---- helpers -------------------------------------------------------------
  auto loadChunk = [&](v16h (&b)[8], const _Float16* wseg, int chanElemOff) {
#pragma unroll
    for (int t = 0; t < 8; ++t)
      b[t] = *(const v16h*)(wseg + boff[t] + chanElemOff);
  };
  auto wmmaChunk = [&](const v16h& a, const v16h (&b)[8]) {
#pragma unroll
    for (int t = 0; t < 8; ++t)
      acc[t] = __builtin_amdgcn_wmma_f32_16x16x32_f16(
          false, a, false, b[t], (short)0, acc[t], false, false);
  };
  auto packA = [&](const v8f& va, const v8f& vb) -> v16h {
    v16h a;
#pragma unroll
    for (int h = 0; h < 8; ++h) {
      a[h]     = (_Float16)va[h];
      a[h + 8] = (_Float16)vb[h];
    }
    return a;
  };
  // --------------------------------------------------------------------------

#pragma unroll 1
  for (int i0 = 0; i0 < IN_DIM; i0 += 32) {
    const _Float16* wseg = Wh + i0;              // uniform (SGPR) segment base

    // prologue: start channel-0 B loads before any compute of this i-block
    loadChunk(bf[0], wseg, 0 * IN_DIM);

    // 16 x-values per lane, reused for all 17 channels
    const float* xseg = x + i0;                  // uniform base
    v8f xa = *(const v8f*)(xseg + xoff);         // K = ka0 .. ka0+7
    v8f xb = *(const v8f*)(xseg + xoff + 8);     // K = ka0+8 .. ka0+15

    // channel 0: silu(x) = x * rcp(1 + exp(-x))  (no fp32 divide)
    v8f sva, svb;
#pragma unroll
    for (int h = 0; h < 8; ++h) {
      float xv = xa[h], xw = xb[h];
      sva[h] = xv * __builtin_amdgcn_rcpf(1.0f + __expf(-xv));
      svb[h] = xw * __builtin_amdgcn_rcpf(1.0f + __expf(-xw));
    }
    v16h a0 = packA(sva, svb);

    loadChunk(bf[1], wseg, 1 * IN_DIM);          // prefetch cos1
    wmmaChunk(a0, bf[0]);                        // consume channel 0

    // base harmonics: the ONLY trig in the i-block (2 TRANS / value)
    v8f c1a, s1a, c1b, s1b;
#pragma unroll
    for (int h = 0; h < 8; ++h) {
      float ta = xa[h] * INV_2PI;
      float tb = xb[h] * INV_2PI;
      c1a[h] = __builtin_amdgcn_cosf(ta);
      s1a[h] = __builtin_amdgcn_sinf(ta);
      c1b[h] = __builtin_amdgcn_cosf(tb);
      s1b[h] = __builtin_amdgcn_sinf(tb);
    }
    v8f cka = c1a, ska = s1a, ckb = c1b, skb = s1b;   // cos(kx), sin(kx)

#pragma unroll
    for (int k = 1; k <= GSIZE; ++k) {
      // cos_k is resident in bf[k&1]; prefetch sin_k into the other buffer
      loadChunk(bf[(k + 1) & 1], wseg, (GSIZE + k) * IN_DIM);
      v16h ac = packA(cka, ckb);
      wmmaChunk(ac, bf[k & 1]);

      // prefetch cos_{k+1} while sin_k's WMMAs run
      if (k < GSIZE) loadChunk(bf[k & 1], wseg, (k + 1) * IN_DIM);
      v16h as = packA(ska, skb);
      wmmaChunk(as, bf[(k + 1) & 1]);

      // angle-addition step (packed FMAs, dual-issue under WMMA shadow):
      //   cos((k+1)x) = cos(kx)cos(x) - sin(kx)sin(x)
      //   sin((k+1)x) = sin(kx)cos(x) + cos(kx)sin(x)
      if (k < GSIZE) {
        v8f cna = cka * c1a - ska * s1a;
        v8f sna = ska * c1a + cka * s1a;
        v8f cnb = ckb * c1b - skb * s1b;
        v8f snb = skb * c1b + ckb * s1b;
        cka = cna; ska = sna; ckb = cnb; skb = snb;
      }
    }
  }

  // C/D layout: VGPR r -> M = r (lanes 0-15) or 8+r (lanes 16-31); N = lane15
  const int orow = m0 + (hi ? 8 : 0);
#pragma unroll
  for (int t = 0; t < 8; ++t) {
    float* op = out + (size_t)orow * OUT_DIM + (n0 + t * 16 + lane15);
#pragma unroll
    for (int r = 0; r < 8; ++r)
      op[(size_t)r * OUT_DIM] = acc[t][r];
  }
}

extern "C" void kernel_launch(void* const* d_in, const int* in_sizes, int n_in,
                              void* d_out, int out_size, void* d_ws, size_t ws_size,
                              hipStream_t stream) {
  const float* x            = (const float*)d_in[0];
  const float* scale_base   = (const float*)d_in[1];
  const float* scale_spline = (const float*)d_in[2];
  const float* coeff        = (const float*)d_in[3];
  float* out                = (float*)d_out;
  _Float16* Wh              = (_Float16*)d_ws;   // 1024*17408*2B = 34 MB

  // Pass 1: pack WMMA-ready fp16 weights into workspace
  kan_pack_weights<<<(OUT_DIM * IN_DIM) / 256, 256, 0, stream>>>(
      scale_base, scale_spline, coeff, Wh);

  // Pass 2: fused feature-gen + WMMA GEMM
  dim3 grid(OUT_DIM / 128, BATCH / 128);   // (8, 32)
  kan_fft_wmma<<<grid, 256, 0, stream>>>(x, Wh, out);
}
